// PlacementPredictor_86320252715187
// MI455X (gfx1250) — matrix operand order repaired and checked by reading, not compiled, not run on hardware
//
#include <hip/hip_runtime.h>
#include <hip/hip_bf16.h>
#include <stdint.h>

typedef __bf16 bf16_t;
typedef bf16_t v16bf __attribute__((ext_vector_type(16)));
typedef bf16_t v8bf  __attribute__((ext_vector_type(8)));
typedef float  v8f   __attribute__((ext_vector_type(8)));

#define NEG_SLOPE 0.01f

__device__ __forceinline__ bf16_t f2bf(float x) {
  unsigned u = __builtin_bit_cast(unsigned, x);
  u += 0x7FFFu + ((u >> 16) & 1u);               // round-to-nearest-even
  unsigned short h = (unsigned short)(u >> 16);
  return __builtin_bit_cast(bf16_t, h);
}

// monotonic float<->uint encoding for atomicMax on floats
__device__ __forceinline__ unsigned encf(float x) {
  unsigned u = __builtin_bit_cast(unsigned, x);
  return u ^ (unsigned)(((int)u >> 31) | (int)0x80000000u);
}
__device__ __forceinline__ float decf(unsigned e) {
  unsigned u = e ^ ((e & 0x80000000u) ? 0x80000000u : 0xFFFFFFFFu);
  return __builtin_bit_cast(float, u);
}

// ---------------------------------------------------------------------------
// cast f32 [rows x cols] -> bf16 [rows x colsPad], zero padding
__global__ void k_cast_pad(const float* __restrict__ X, bf16_t* __restrict__ Y,
                           long rows, int cols, int colsPad) {
  long idx = blockIdx.x * (long)blockDim.x + threadIdx.x;
  long total = rows * (long)colsPad;
  if (idx >= total) return;
  int  c = (int)(idx % colsPad);
  long r = idx / colsPad;
  float v = (c < cols) ? X[r * (long)cols + c] : 0.0f;
  Y[idx] = f2bf(v);
}

// pack W [K x do] (row-major f32) into per-wave B fragments for wmma bf16:
// Bp[((nt*nKt + kt)*32 + lane)*16 + i] = W[kt*32 + (lane>>4)*8 + (i&7) + (i>>3)*16][nt*16 + (lane&15)]
__global__ void k_pack_w(const float* __restrict__ W, bf16_t* __restrict__ Bp,
                         int K, int dO, int Kpad, int Npad) {
  int idx = blockIdx.x * blockDim.x + threadIdx.x;
  int nKt = Kpad >> 5;
  int total = (Npad >> 4) * nKt * 512;
  if (idx >= total) return;
  int i    = idx & 15;
  int lane = (idx >> 4) & 31;
  int kt   = (idx >> 9) % nKt;
  int nt   = (idx >> 9) / nKt;
  int k = kt * 32 + (lane >> 4) * 8 + (i & 7) + (i >> 3) * 16;
  int n = nt * 16 + (lane & 15);
  float v = (k < K && n < dO) ? W[k * dO + n] : 0.0f;
  Bp[idx] = f2bf(v);
}

// C[M x dO] = A[M x (NK*32)] * Bpacked (+bias).
// One wave computes a 16-row x (NT*16)-col stripe: A fragment loaded once per
// K-step, reused across NT WMMAs (B fragments are L0-resident weights).
template <int NT, int NK>
__global__ __launch_bounds__(256)
void k_gemm_bf16(const bf16_t* __restrict__ A, const bf16_t* __restrict__ Bp,
                 float* __restrict__ C, const float* __restrict__ bias,
                 int M, int dO) {
  const int Kpad = NK * 32;
  int wave = (blockIdx.x * blockDim.x + threadIdx.x) >> 5;
  int lane = threadIdx.x & 31;
  int mtile = wave;
  if (mtile * 16 >= M) return;

  int m  = mtile * 16 + (lane & 15);
  int kh = lane >> 4;                 // 0: K 0-7/16-23, 1: K 8-15/24-31

  const bf16_t* arow = A + (size_t)m * Kpad + kh * 8;
  const bf16_t* bp   = Bp + (size_t)lane * 16;

  v8f acc[NT] = {};
#pragma unroll
  for (int kt = 0; kt < NK; ++kt) {
    v8bf alo = *(const v8bf*)(arow + kt * 32);
    v8bf ahi = *(const v8bf*)(arow + kt * 32 + 16);
    v16bf a;
#pragma unroll
    for (int i = 0; i < 8; ++i) { a[i] = alo[i]; a[8 + i] = ahi[i]; }
#pragma unroll
    for (int nt = 0; nt < NT; ++nt) {
      v16bf b = *(const v16bf*)(bp + (size_t)(nt * NK + kt) * 512);
      acc[nt] = __builtin_amdgcn_wmma_f32_16x16x32_bf16(
          false, a, false, b, (short)0, acc[nt], false, false);
    }
  }

  int nlane = lane & 15;
  int mbase = mtile * 16 + kh * 8;
#pragma unroll
  for (int nt = 0; nt < NT; ++nt) {
    int n = nt * 16 + nlane;
    if (n < dO) {
      float bv = bias ? bias[n] : 0.0f;
#pragma unroll
      for (int r = 0; r < 8; ++r)
        C[(size_t)(mbase + r) * dO + n] = acc[nt][r] + bv;
    }
  }
}

// f = leaky_relu(P[src] + Q[dst] + f + bias); optionally score = dot(f, attn)
// one wave per edge; runs in place over the edge-GEMM output
__global__ __launch_bounds__(256)
void k_edge_combine(float* __restrict__ F,
                    const float* __restrict__ P, const float* __restrict__ Q,
                    const int* __restrict__ src, const int* __restrict__ dst,
                    const float* __restrict__ bias, const float* __restrict__ attn,
                    float* __restrict__ score, long E, int dO) {
  long wid = (blockIdx.x * (long)blockDim.x + threadIdx.x) >> 5;
  int lane = threadIdx.x & 31;
  if (wid >= E) return;
  int si = src[wid], di = dst[wid];
  const float* p = P + (size_t)si * dO;
  const float* q = Q + (size_t)di * dO;
  float*       f = F + (size_t)wid * dO;
  float acc = 0.0f;
  for (int c = lane; c < dO; c += 32) {
    float v = p[c] + q[c] + f[c] + bias[c];
    v = (v > 0.0f) ? v : NEG_SLOPE * v;
    f[c] = v;
    if (attn) acc += v * attn[c];
  }
  if (score) {
#pragma unroll
    for (int off = 16; off > 0; off >>= 1) acc += __shfl_xor(acc, off, 32);
    if (lane == 0) score[wid] = acc;
  }
}

__global__ void k_fill_f32(float* p, float v, long n) {
  long i = blockIdx.x * (long)blockDim.x + threadIdx.x;
  if (i < n) p[i] = v;
}
__global__ void k_fill_u32(unsigned* p, unsigned v, long n) {
  long i = blockIdx.x * (long)blockDim.x + threadIdx.x;
  if (i < n) p[i] = v;
}

__global__ void k_seg_max(const float* __restrict__ score, const int* __restrict__ dst,
                          unsigned* __restrict__ menc, long E) {
  long e = blockIdx.x * (long)blockDim.x + threadIdx.x;
  if (e >= E) return;
  atomicMax(menc + dst[e], encf(score[e]));
}

__global__ void k_edge_exp(const float* __restrict__ score, const int* __restrict__ dst,
                           const unsigned* __restrict__ menc,
                           float* __restrict__ ex, float* __restrict__ ssum, long E) {
  long e = blockIdx.x * (long)blockDim.x + threadIdx.x;
  if (e >= E) return;
  int d = dst[e];
  float v = expf(score[e] - decf(menc[d]));
  ex[e] = v;
  atomicAdd(ssum + d, v);
}

// h_out[dst] += H[src] * (ex/ssum[dst]) ; one wave per edge
__global__ __launch_bounds__(256)
void k_norm_agg(const float* __restrict__ ex, const float* __restrict__ ssum,
                const float* __restrict__ H, const int* __restrict__ src,
                const int* __restrict__ dst, float* __restrict__ hout,
                long E, int dO) {
  long wid = (blockIdx.x * (long)blockDim.x + threadIdx.x) >> 5;
  int lane = threadIdx.x & 31;
  if (wid >= E) return;
  int si = src[wid], di = dst[wid];
  float a = ex[wid] / ssum[di];
  const float* hs = H + (size_t)si * dO;
  float*       hd = hout + (size_t)di * dO;
  for (int c = lane; c < dO; c += 32)
    atomicAdd(hd + c, hs[c] * a);
}

// ---------------------------------------------------------------------------
static inline unsigned cdl(long a, long b) { return (unsigned)((a + b - 1) / b); }

static void launch_gemm(const bf16_t* A, const bf16_t* Bp, float* C, const float* bias,
                        int M, int Kpad, int Npad, int dO, hipStream_t stream) {
  unsigned blocks = cdl((long)(M / 16) * 32, 256);
  if (Kpad == 128 && Npad == 128)
    k_gemm_bf16<8, 4><<<blocks, 256, 0, stream>>>(A, Bp, C, bias, M, dO);
  else if (Kpad == 128 && Npad == 32)
    k_gemm_bf16<2, 4><<<blocks, 256, 0, stream>>>(A, Bp, C, bias, M, dO);
  else if (Kpad == 32 && Npad == 64)
    k_gemm_bf16<4, 1><<<blocks, 256, 0, stream>>>(A, Bp, C, bias, M, dO);
}

extern "C" void kernel_launch(void* const* d_in, const int* in_sizes, int n_in,
                              void* d_out, int out_size, void* d_ws, size_t ws_size,
                              hipStream_t stream) {
  (void)in_sizes; (void)n_in; (void)out_size; (void)ws_size;
  const int  N = 50000;
  const long E = 800000;

  const float* x_in = (const float*)d_in[0];
  const float* e_in = (const float*)d_in[1];
  const int*   src  = (const int*)d_in[2];
  const int*   dst  = (const int*)d_in[3];
  const float* W_n[3]  = {(const float*)d_in[4],  (const float*)d_in[11], (const float*)d_in[18]};
  const float* b_n[3]  = {(const float*)d_in[5],  (const float*)d_in[12], (const float*)d_in[19]};
  const float* W_ni[3] = {(const float*)d_in[6],  (const float*)d_in[13], (const float*)d_in[20]};
  const float* W_nj[3] = {(const float*)d_in[7],  (const float*)d_in[14], (const float*)d_in[21]};
  const float* W_f[3]  = {(const float*)d_in[8],  (const float*)d_in[15], (const float*)d_in[22]};
  const float* attn[3] = {(const float*)d_in[9],  (const float*)d_in[16], (const float*)d_in[23]};
  const float* bias[3] = {(const float*)d_in[10], (const float*)d_in[17], (const float*)d_in[24]};

  const int Kdim[3] = {128, 128, 30};
  const int Kpad[3] = {128, 128, 32};
  const int Od[3]   = {128, 30, 64};
  const int Npad[3] = {128, 32, 64};

  // workspace carve-up (256B aligned regions)
  char* w = (char*)d_ws;
  auto carve = [&](size_t bytes) -> void* {
    void* p = (void*)w;
    w += (bytes + 255) & ~(size_t)255;
    return p;
  };
  bf16_t*  xb    = (bf16_t*)carve((size_t)N * 128 * 2);
  bf16_t*  eb    = (bf16_t*)carve((size_t)E * 128 * 2);
  float*   P     = (float*)carve((size_t)N * 128 * 4);
  float*   Q     = (float*)carve((size_t)N * 128 * 4);
  float*   H     = (float*)carve((size_t)N * 128 * 4);
  float*   hA    = (float*)carve((size_t)N * 128 * 4);   // h_out layer 1
  float*   hB    = (float*)carve((size_t)N * 128 * 4);   // h_out layer 2
  float*   R0    = (float*)carve((size_t)E * 128 * 4);   // f1 [E x 128]
  float*   R1    = (float*)carve((size_t)E * 32 * 4);    // f2 [E x 30]
  float*   score = (float*)carve((size_t)E * 4);
  float*   exb   = (float*)carve((size_t)E * 4);
  unsigned* menc = (unsigned*)carve((size_t)N * 4);
  float*   ssum  = (float*)carve((size_t)N * 4);
  bf16_t*  wpNI  = (bf16_t*)carve(65536);
  bf16_t*  wpNJ  = (bf16_t*)carve(65536);
  bf16_t*  wpF   = (bf16_t*)carve(65536);
  bf16_t*  wpN   = (bf16_t*)carve(65536);

  const unsigned ENC_NEG_INF = 0x007FFFFFu;  // encf(-inf)

  for (int l = 0; l < 3; ++l) {
    const float* xcur = (l == 0) ? x_in : (l == 1 ? hA : hB);
    const float* ecur = (l == 0) ? e_in : (l == 1 ? R0 : R1);
    float*       fout = (l == 0) ? R0 : (l == 1 ? R1 : (float*)d_out);
    float*       hnext = (l == 0) ? hA : hB;

    // 1) cast inputs to padded bf16
    long tx = (long)N * Kpad[l];
    k_cast_pad<<<cdl(tx, 256), 256, 0, stream>>>(xcur, xb, N, Kdim[l], Kpad[l]);
    long te = E * (long)Kpad[l];
    k_cast_pad<<<cdl(te, 256), 256, 0, stream>>>(ecur, eb, E, Kdim[l], Kpad[l]);

    // 2) pack weights into wmma B fragments
    int ptot = (Npad[l] >> 4) * (Kpad[l] >> 5) * 512;
    k_pack_w<<<cdl(ptot, 256), 256, 0, stream>>>(W_ni[l], wpNI, Kdim[l], Od[l], Kpad[l], Npad[l]);
    k_pack_w<<<cdl(ptot, 256), 256, 0, stream>>>(W_nj[l], wpNJ, Kdim[l], Od[l], Kpad[l], Npad[l]);
    k_pack_w<<<cdl(ptot, 256), 256, 0, stream>>>(W_f[l],  wpF,  Kdim[l], Od[l], Kpad[l], Npad[l]);
    if (l < 2)
      k_pack_w<<<cdl(ptot, 256), 256, 0, stream>>>(W_n[l], wpN, Kdim[l], Od[l], Kpad[l], Npad[l]);

    // 3) GEMMs (WMMA bf16 -> f32)
    launch_gemm(xb, wpNI, P, nullptr, N, Kpad[l], Npad[l], Od[l], stream);
    launch_gemm(xb, wpNJ, Q, nullptr, N, Kpad[l], Npad[l], Od[l], stream);
    if (l < 2)
      launch_gemm(xb, wpN, H, b_n[l], N, Kpad[l], Npad[l], Od[l], stream);
    launch_gemm(eb, wpF, fout, nullptr, (int)E, Kpad[l], Npad[l], Od[l], stream);

    // 4) fused edge combine (+score for layers 1-2), in place over fout
    bool needScore = (l < 2);
    k_edge_combine<<<cdl(E * 32, 256), 256, 0, stream>>>(
        fout, P, Q, src, dst, bias[l],
        needScore ? attn[l] : nullptr, needScore ? score : nullptr, E, Od[l]);

    if (l < 2) {
      // 5) segmented softmax over dst + attention-weighted aggregation
      k_fill_u32<<<cdl(N, 256), 256, 0, stream>>>(menc, ENC_NEG_INF, N);
      k_fill_f32<<<cdl(N, 256), 256, 0, stream>>>(ssum, 0.0f, N);
      k_fill_f32<<<cdl((long)N * Od[l], 256), 256, 0, stream>>>(hnext, 0.0f, (long)N * Od[l]);
      k_seg_max<<<cdl(E, 256), 256, 0, stream>>>(score, dst, menc, E);
      k_edge_exp<<<cdl(E, 256), 256, 0, stream>>>(score, dst, menc, exb, ssum, E);
      k_norm_agg<<<cdl(E * 32, 256), 256, 0, stream>>>(exb, ssum, H, src, dst, hnext, E, Od[l]);
    }
  }
}
